// SpMiddleFHD_84146999263311
// MI455X (gfx1250) — matrix operand order repaired and checked by reading, not compiled, not run on hardware
//
#include <hip/hip_runtime.h>

// ---------------------------------------------------------------------------
// SpMiddleFHD (spconv middle extractor) for MI455X / gfx1250.
// Active-site compaction + gather implicit-GEMM on f16 WMMA
// (v_wmma_f32_16x16x32_f16), dense zero-filled f16 feature buffers so border /
// inactive gathers are trivially correct, fused BN+ReLU epilogue.
// Conv kernels are template-specialized per (cin, taps, cout-tiles) so the
// K-chunk loop is compile-time and the WMMA chain can be unrolled/pipelined.
// Assumption: the bool mask input is a 1-byte-per-element buffer.
// ---------------------------------------------------------------------------

typedef __attribute__((ext_vector_type(16))) _Float16 v16h;
typedef __attribute__((ext_vector_type(8)))  float    v8f;

// ---------------------------------------------------------------- compaction
__global__ void compact_mask(const unsigned char* __restrict__ m,
                             int* __restrict__ list, int* __restrict__ counter, int n) {
  int i = blockIdx.x * blockDim.x + threadIdx.x;
  if (i >= n) return;
  if (m[i]) {
    int k = atomicAdd(counter, 1);
    list[k] = i;
  }
}

// ----------------------------------------------- input NCDHW f32 -> f16 CL
__global__ void convert_input(const float* __restrict__ x,
                              const unsigned char* __restrict__ m,
                              _Float16* __restrict__ f, int n) {
  int i = blockIdx.x * blockDim.x + threadIdx.x;
  if (i >= n) return;
  bool a = m[i] != 0;
#pragma unroll
  for (int c = 0; c < 16; ++c)
    f[(size_t)i * 16 + c] = a ? (_Float16)x[(size_t)c * n + i] : (_Float16)0.f;
}

// ------------------------------------- weights -> WMMA A-fragment lane order
// A (16x32 f16, MxK): lane l: m=l&15, half=l>>4; elem i: v=i>>1, odd=i&1;
//   k = v<4 ? 2v+8*half+odd : 16+2(v-4)+8*half+odd.   K order = tap*cin+ci.
__global__ void pack_weights(const float* __restrict__ w, _Float16* __restrict__ out,
                             int cout, int cin, int taps, int KT, int total) {
  int idx = blockIdx.x * blockDim.x + threadIdx.x;
  if (idx >= total) return;
  int i = idx & 15, frag = idx >> 4;
  int lane = frag & 31, blk = frag >> 5;
  int chunk = blk % KT, mt = blk / KT;
  int half = lane >> 4, m = mt * 16 + (lane & 15);
  int v = i >> 1, odd = i & 1;
  int kk = (v < 4) ? (2 * v + 8 * half + odd) : (16 + 2 * (v - 4) + 8 * half + odd);
  int K = chunk * 32 + kk;
  float val = 0.f;
  if (m < cout && K < cin * taps) {
    int tap = K / cin, ci = K - tap * cin;
    val = w[((size_t)m * cin + ci) * taps + tap];
  }
  out[idx] = (_Float16)val;
}

// ------------------------------------ spconv mask: ones-conv > 0, + compact
__global__ void spconv_mask(const unsigned char* __restrict__ inM,
                            unsigned char* __restrict__ outM,
                            int* __restrict__ list, int* __restrict__ counter,
                            int ID, int IH, int IW, int OHW, int OW, int nOut,
                            int kd, int kh, int kw, int sd, int sh, int sw,
                            int pd, int ph, int pw, int dd, int dh, int dw) {
  int i = blockIdx.x * blockDim.x + threadIdx.x;
  if (i >= nOut) return;
  int od = i / OHW, r = i - od * OHW, oh = r / OW, ow = r - oh * OW;
  int act = 0;
  for (int kz = 0; kz < kd && !act; ++kz) {
    int id = od * sd - pd + kz * dd;
    if ((unsigned)id >= (unsigned)ID) continue;
    for (int ky = 0; ky < kh && !act; ++ky) {
      int ih = oh * sh - ph + ky * dh;
      if ((unsigned)ih >= (unsigned)IH) continue;
      for (int kx = 0; kx < kw; ++kx) {
        int iw = ow * sw - pw + kx * dw;
        if ((unsigned)iw >= (unsigned)IW) continue;
        if (inM[((size_t)id * IH + ih) * IW + iw]) { act = 1; break; }
      }
    }
  }
  outM[i] = (unsigned char)act;
  if (act) {
    int k = atomicAdd(counter, 1);
    list[k] = i;
  }
}

// -------------------------------------------------- gather implicit-GEMM conv
// Block = 32*MT threads, owns a tile of 16 active output voxels. All threads
// stage the 16 gathered patches (TAPS*CIN f16, zero-padded) into LDS once;
// wave w computes cout tile w with a compile-time chain of
// v_wmma_f32_16x16x32_f16 ops over KT = KPAD/32 chunks.
template <int CIN, int TAPS, int MT, bool FINAL>
__global__ __launch_bounds__(32 * MT)
void conv_wmma(const _Float16* __restrict__ inF, _Float16* __restrict__ outF,
               float* __restrict__ outFinal,
               const int* __restrict__ list, const int* __restrict__ counter,
               const _Float16* __restrict__ Apack,
               const float* __restrict__ gma, const float* __restrict__ bta,
               int ID, int IH, int IW, int OH, int OW,
               int sd, int sh, int sw, int pd, int ph, int pw,
               int dd, int dh, int dw, int kh, int kw) {
  constexpr int KPAD  = ((CIN * TAPS + 31) / 32) * 32;
  constexpr int KT    = KPAD / 32;
  constexpr int PITCH = KPAD + 8;        // +16B => conflict-free ds_read_b128
  constexpr int NTHR  = 32 * MT;

  extern __shared__ __align__(128) unsigned char smem[];
  int* vox = (int*)smem;
  _Float16* patch = (_Float16*)(smem + 64);

  int count = counter[0];
  int n0 = blockIdx.x * 16;
  if (n0 >= count) return;               // uniform early-exit for idle tiles

  int tid = threadIdx.x;
  if (tid < 16) vox[tid] = (n0 + tid < count) ? list[n0 + tid] : -1;
  __syncthreads();

  const int OHW = OH * OW;
  const int khw = kh * kw;
  constexpr int vecsPerTap = CIN / 8;    // CIN f16 in 16B chunks
  const uint4 zero4 = {0u, 0u, 0u, 0u};

  for (int e = tid; e < 16 * TAPS; e += NTHR) {
    int n = e / TAPS, tap = e - n * TAPS;
    uint4* dst = (uint4*)(patch + n * PITCH + tap * CIN);
    int v = vox[n];
    const uint4* src = nullptr;
    if (v >= 0) {
      int od = v / OHW, r2 = v - od * OHW, oh = r2 / OW, ow = r2 - oh * OW;
      int kz = tap / khw, rr = tap - kz * khw, ky = rr / kw, kx = rr - ky * kw;
      int id = od * sd - pd + kz * dd;
      int ih = oh * sh - ph + ky * dh;
      int iw = ow * sw - pw + kx * dw;
      if ((unsigned)id < (unsigned)ID && (unsigned)ih < (unsigned)IH &&
          (unsigned)iw < (unsigned)IW)
        src = (const uint4*)(inF + ((size_t)(id * IH + ih) * IW + iw) * CIN);
    }
    if (src) {
#pragma unroll
      for (int j = 0; j < vecsPerTap; ++j) dst[j] = src[j];
    } else {
#pragma unroll
      for (int j = 0; j < vecsPerTap; ++j) dst[j] = zero4;
    }
  }
  constexpr int padVecs = (KPAD - CIN * TAPS) / 8;  // K padding to mult of 32
  if constexpr (padVecs > 0) {
    for (int e = tid; e < 16 * padVecs; e += NTHR) {
      int n = e / padVecs, j = e - n * padVecs;
      ((uint4*)(patch + n * PITCH + CIN * TAPS))[j] = zero4;
    }
  }
  __syncthreads();

  int w = tid >> 5, lane = tid & 31;
  int nl = lane & 15, half = lane >> 4;
  v8f acc = {};
  const v16h* A = ((const v16h*)Apack) + (size_t)w * KT * 32;  // lane-packed
  const _Float16* rowB = patch + nl * PITCH + half * 8;        // table-order K

#pragma unroll 2
  for (int c = 0; c < KT; ++c) {
    v16h a = A[c * 32 + lane];                    // one 32B coalesced load
    if (c + 1 < KT) __builtin_prefetch(&A[(c + 1) * 32 + lane], 0, 3);
    union { uint4 u[2]; v16h v; } bu;             // 2x b128 LDS reads
    bu.u[0] = *(const uint4*)(rowB + c * 32);
    bu.u[1] = *(const uint4*)(rowB + c * 32 + 16);
    acc = __builtin_amdgcn_wmma_f32_16x16x32_f16(
        false, a, false, bu.v, (short)0, acc, false, false);
  }

  int v = vox[nl];
  if (v < 0) return;
#pragma unroll
  for (int r = 0; r < 8; ++r) {                   // C/D layout: m = r+8*half
    int m = (w << 4) + (half << 3) + r;
    float val = acc[r] * gma[m] + bta[m];         // folded BN
    val = val > 0.f ? val : 0.f;                  // ReLU
    if constexpr (FINAL) outFinal[(size_t)m * OHW + v] = val;
    else                 outF[(size_t)v * (16 * MT) + m] = (_Float16)val;
  }
}

// ---------------------------------------------------------------------------
static inline size_t alignUp(size_t x) { return (x + 255) & ~(size_t)255; }

extern "C" void kernel_launch(void* const* d_in, const int* in_sizes, int n_in,
                              void* d_out, int out_size, void* d_ws, size_t ws_size,
                              hipStream_t stream) {
  (void)in_sizes; (void)n_in;
  const float* x = (const float*)d_in[0];
  const unsigned char* mask0 = (const unsigned char*)d_in[1];
  const float *Wt[14], *Gs[14], *Bs[14];
  for (int i = 0; i < 14; ++i) {
    Wt[i] = (const float*)d_in[2 + i];
    Gs[i] = (const float*)d_in[16 + i];
    Bs[i] = (const float*)d_in[30 + i];
  }

  // Level geometry: L0..L4
  static const int LDv[5] = {25, 13, 7, 3, 1};
  static const int LHv[5] = {200, 100, 50, 25, 25};
  static const int LWv[5] = {176, 88, 44, 22, 22};
  int vox[5];
  for (int l = 0; l < 5; ++l) vox[l] = LDv[l] * LHv[l] * LWv[l];

  // Layer table
  static const int T[14]    = {0,0,1,0,0,1,0,0,0,1,0,0,0,1};
  static const int CIN[14]  = {16,16,16,32,32,32,64,64,64,64,64,64,64,64};
  static const int COUT[14] = {16,16,32,32,32,64,64,64,64,64,64,64,64,64};
  static const int KD[14]   = {3,3,3,3,3,3,3,3,3,3,3,3,3,3};
  static const int KH[14]   = {3,3,3,3,3,3,3,3,3,3,3,3,3,1};
  static const int KW[14]   = {3,3,3,3,3,3,3,3,3,3,3,3,3,1};
  static const int SD[14]   = {1,1,2,1,1,2,1,1,1,2,1,1,1,2};
  static const int SH[14]   = {1,1,2,1,1,2,1,1,1,2,1,1,1,1};
  static const int SW[14]   = {1,1,2,1,1,2,1,1,1,2,1,1,1,1};
  static const int PD[14]   = {1,2,1,1,2,1,1,2,3,0,1,2,3,0};
  static const int PH[14]   = {1,2,1,1,2,1,1,2,3,1,1,2,3,0};
  static const int PW[14]   = {1,2,1,1,2,1,1,2,3,1,1,2,3,0};
  static const int DIL[14]  = {1,2,1,1,2,1,1,2,3,1,1,2,3,1};
  static const int INL[14]  = {0,0,0,1,1,1,2,2,2,2,3,3,3,3};
  static const int OUTL[14] = {0,0,1,1,1,2,2,2,2,3,3,3,3,4};

  // Workspace layout
  size_t off = 0;
  auto take = [&](size_t bytes) { size_t o = off; off = alignUp(off + bytes); return o; };
  size_t featBytes = (size_t)vox[0] * 16 * 2;          // max feature buffer
  size_t oFA = take(featBytes), oFB = take(featBytes);
  size_t oMask[5], oList[5];
  for (int l = 0; l < 5; ++l) {
    oMask[l] = take((size_t)vox[l]);
    oList[l] = take((size_t)vox[l] * 4);
  }
  size_t oCnt = take(5 * sizeof(int));
  size_t oW[14];
  int KpadL[14], tapsL[14];
  for (int i = 0; i < 14; ++i) {
    tapsL[i] = KD[i] * KH[i] * KW[i];
    KpadL[i] = ((CIN[i] * tapsL[i] + 31) / 32) * 32;
    size_t elems = (size_t)(COUT[i] / 16) * (KpadL[i] / 32) * 512;
    oW[i] = take(elems * 2);
  }
  if (off > ws_size) return;  // insufficient workspace (deterministic no-op)

  char* W = (char*)d_ws;
  int* counters = (int*)(W + oCnt);

  hipMemsetAsync(counters, 0, 5 * sizeof(int), stream);

  // Level-0 active list + f16 channels-last masked features
  compact_mask<<<(vox[0] + 255) / 256, 256, 0, stream>>>(
      mask0, (int*)(W + oList[0]), counters + 0, vox[0]);
  convert_input<<<(vox[0] + 255) / 256, 256, 0, stream>>>(
      x, mask0, (_Float16*)(W + oFA), vox[0]);

  // Pack all layer weights into WMMA A-fragment order (f16)
  for (int i = 0; i < 14; ++i) {
    int KT = KpadL[i] / 32;
    int total = (COUT[i] / 16) * KT * 512;
    pack_weights<<<(total + 255) / 256, 256, 0, stream>>>(
        Wt[i], (_Float16*)(W + oW[i]), COUT[i], CIN[i], tapsL[i], KT, total);
  }

  const unsigned char* maskPtr[5];
  maskPtr[0] = mask0;
  for (int l = 1; l < 5; ++l) maskPtr[l] = (const unsigned char*)(W + oMask[l]);

  _Float16* fIn  = (_Float16*)(W + oFA);
  _Float16* fOut = (_Float16*)(W + oFB);

  for (int i = 0; i < 14; ++i) {
    int iL = INL[i], oL = OUTL[i];
    if (T[i]) {  // spconv: grow mask, build new active list at out level
      spconv_mask<<<(vox[oL] + 255) / 256, 256, 0, stream>>>(
          maskPtr[iL], (unsigned char*)(W + oMask[oL]),
          (int*)(W + oList[oL]), counters + oL,
          LDv[iL], LHv[iL], LWv[iL], LHv[oL] * LWv[oL], LWv[oL], vox[oL],
          KD[i], KH[i], KW[i], SD[i], SH[i], SW[i],
          PD[i], PH[i], PW[i], DIL[i], DIL[i], DIL[i]);
    }
    bool fin = (i == 13);
    if (fin) hipMemsetAsync(d_out, 0, (size_t)out_size * 4, stream);
    else     hipMemsetAsync(fOut, 0, (size_t)vox[oL] * COUT[i] * 2, stream);

    int MT = COUT[i] / 16;
    size_t shmem = 64 + (size_t)16 * (KpadL[i] + 8) * 2;
    int tiles = (vox[oL] + 15) / 16;

#define CONV_ARGS                                                            \
    fIn, fOut, (float*)d_out, (int*)(W + oList[oL]), counters + oL,          \
    (const _Float16*)(W + oW[i]), Gs[i], Bs[i],                              \
    LDv[iL], LHv[iL], LWv[iL], LHv[oL], LWv[oL],                             \
    SD[i], SH[i], SW[i], PD[i], PH[i], PW[i],                                \
    DIL[i], DIL[i], DIL[i], KH[i], KW[i]

    if (i <= 1) {              // subm 16->16
      conv_wmma<16, 27, 1, false><<<tiles, 32, shmem, stream>>>(CONV_ARGS);
    } else if (i == 2) {       // spconv 16->32
      conv_wmma<16, 27, 2, false><<<tiles, 64, shmem, stream>>>(CONV_ARGS);
    } else if (i <= 4) {       // subm 32->32
      conv_wmma<32, 27, 2, false><<<tiles, 64, shmem, stream>>>(CONV_ARGS);
    } else if (i == 5) {       // spconv 32->64
      conv_wmma<32, 27, 4, false><<<tiles, 128, shmem, stream>>>(CONV_ARGS);
    } else if (i <= 12) {      // subm/spconv 64->64, 27 taps
      conv_wmma<64, 27, 4, false><<<tiles, 128, shmem, stream>>>(CONV_ARGS);
    } else {                   // final spconv 64->64, k=(3,1,1), f32 out
      conv_wmma<64, 3, 4, true><<<tiles, 128, shmem, stream>>>(CONV_ARGS);
    }
#undef CONV_ARGS

    _Float16* t = fIn; fIn = fOut; fOut = t;
  }
}
